// RegionProposalNetwork_71897752535536
// MI455X (gfx1250) — matrix operand order repaired and checked by reading, not compiled, not run on hardware
//
#include <hip/hip_runtime.h>

// ---------------------------------------------------------------------------
// RPN head fused for gfx1250 (CDNA5, wave32, WMMA + Tensor Data Mover).
// conv3x3(2048->512) + ReLU + conv1x1(512->9/36) + sigmoid/exp anchor decode.
// Implicit GEMM: M=8192 pixels, N=512, K=2048*9, bf16 WMMA w/ f32 accumulate.
// Tiles staged into double-buffered LDS by TDM (tensor_load_to_lds), with
// hardware row padding (64B data + 16B pad) for conflict-free ds_load_b128.
// ---------------------------------------------------------------------------

typedef __attribute__((ext_vector_type(16))) __bf16 v16bf;
typedef __attribute__((ext_vector_type(8)))  float  v8f;
typedef unsigned int v4u __attribute__((ext_vector_type(4)));
typedef int          v4i __attribute__((ext_vector_type(4)));
typedef int          v8i __attribute__((ext_vector_type(8)));

struct alignas(16) q128 { unsigned int w[4]; };
struct Pair128 { q128 a, b; };

__device__ __forceinline__ v16bf ld_frag(const void* p0, const void* p1) {
  Pair128 r;
  r.a = *reinterpret_cast<const q128*>(p0);
  r.b = *reinterpret_cast<const q128*>(p1);
  return __builtin_bit_cast(v16bf, r);
}

// geometry
constexpr int HW    = 32;
constexpr int CIN   = 2048;
constexpr int HID   = 512;
constexpr int STEPS = 9 * 64;     // 9 taps x (2048/32) k-steps

// LDS layout (bytes); rows padded to 80B (A/B) and 1040B (H): conflict-free.
constexpr int AS_STRIDE = 80;
constexpr int BS_STRIDE = 80;
constexpr int AS0_OFF   = 0;      // A buffers: 2 x 32*80  = 2x2560
constexpr int BS0_OFF   = 5120;   // B buffers: 2 x 512*80 = 2x40960
constexpr int H_OFF     = 5120;   // hidden tile aliases B buf 0 (dead after loop)
constexpr int H_STRIDE  = 1040;
constexpr int S_OFF     = 38400;  // 5120 + 32*1040; still inside B buf 0
constexpr int S_STRIDE  = 52;     // floats (48 used)
constexpr int DYN_LDS   = 87040;  // 5120 + 2*40960

// workspace layout (bytes)
constexpr size_t XP_BYTES = (size_t)8 * 34 * 34 * 2048 * 2;   // 37,879,808
constexpr size_t WT_OFF   = XP_BYTES;
constexpr size_t WT_BYTES = (size_t)9 * 512 * 2048 * 2;       // 18,874,368
constexpr size_t WCR_OFF  = WT_OFF + WT_BYTES;

// ------------------------------ prep kernels -------------------------------

__global__ void k_zero_u32(unsigned int* p, int n) {
  int i = blockIdx.x * 256 + threadIdx.x;
  if (i < n) p[i] = 0u;
}

// NCHW f32 -> padded [b][y+1][x+1][c] bf16 via LDS transpose tiles.
__global__ void k_pad_nhwc(const float* __restrict__ fm, __bf16* __restrict__ Xp) {
  __shared__ float tile[64][33];
  const int blk = blockIdx.x;           // 0..255 = (b,y)
  const int b = blk >> 5, y = blk & 31;
  const int t = threadIdx.x;
  const int x  = t & 31, cl = t >> 5;
  const int j  = t & 63, xw = t >> 6;
  for (int c0 = 0; c0 < CIN; c0 += 64) {
#pragma unroll
    for (int i = 0; i < 8; ++i) {
      int r = i * 8 + cl;
      tile[r][x] = fm[(((size_t)(b * CIN + c0 + r)) * HW + y) * HW + x];
    }
    __syncthreads();
#pragma unroll
    for (int i = 0; i < 8; ++i) {
      int x2 = i * 4 + xw;
      Xp[((size_t)((b * 34 + y + 1) * 34 + (x2 + 1)) << 11) + c0 + j] =
          (__bf16)tile[j][x2];
    }
    __syncthreads();
  }
}

// hidden_w [co][ci][ky][kx] f32 -> Wt [tap][co][ci] bf16
__global__ void k_pack_wt(const float* __restrict__ hw, __bf16* __restrict__ Wt) {
  int i = blockIdx.x * 256 + threadIdx.x;        // < 9*512*2048
  int tap = i / (HID * CIN);
  int rem = i - tap * HID * CIN;
  int co = rem >> 11, ci = rem & 2047;
  Wt[i] = (__bf16)hw[(size_t)(co * CIN + ci) * 9 + tap];
}

// conf_w [9][512], reg_w [36][512] f32 -> Wcr [48][512] bf16 (rows 45..47 = 0)
__global__ void k_pack_wcr(const float* __restrict__ cw, const float* __restrict__ rw,
                           __bf16* __restrict__ Wcr) {
  int i = blockIdx.x * 256 + threadIdx.x;        // < 48*512
  int n = i >> 9, c = i & 511;
  float v = 0.0f;
  if (n < 9)       v = cw[n * HID + c];
  else if (n < 45) v = rw[(n - 9) * HID + c];
  Wcr[i] = (__bf16)v;
}

// --------------------------- TDM tile descriptor ---------------------------
// 2-D tile: `rows` rows x 32 bf16, global row stride 2048 elems; LDS rows
// padded by hardware: 16 DWORDs (64B) data then 4 DWORDs (16B) pad -> 80B.

__device__ __forceinline__ void tdm_load_tile(unsigned lds_addr, const void* gptr,
                                              int rows) {
  const unsigned long long ga = (unsigned long long)(uintptr_t)gptr;
  v4u g0;
  g0[0] = 1u;                                   // count=1 (valid user D#)
  g0[1] = lds_addr;                             // LDS byte address
  g0[2] = (unsigned)ga;                         // global addr [31:0]
  g0[3] = (unsigned)((ga >> 32) & 0x01FFFFFFull) | (2u << 30);  // [56:32], type=2
  v8i g1;
  g1[0] = (1 << 16) |                           // data_size = 2 bytes
          (1 << 20) |                           // pad_enable
          (3 << 22) |                           // pad_interval: 16 DWORDs
          (3 << 25);                            // pad_amount:    4 DWORDs
  g1[1] = (int)0xFFFF0000u;                     // abar=0, tensor_dim0 lo16
  g1[2] = (int)0xFFFF7FFFu;                     // tensor_dim0 hi16, dim1 lo16
  g1[3] = (int)(0x7FFFu | (32u << 16));         // dim1 hi16, tile_dim0 = 32
  g1[4] = rows & 0xFFFF;                        // tile_dim1 = rows, tile_dim2 = 0
  g1[5] = 2048;                                 // tensor_dim0_stride lo32 (elems)
  g1[6] = 0;                                    // stride0 hi16, stride1 lo16
  g1[7] = 0;                                    // stride1 hi32
  const v4i z4 = {0, 0, 0, 0};
#if __clang_major__ >= 23
  const v8i z8 = {0, 0, 0, 0, 0, 0, 0, 0};
  __builtin_amdgcn_tensor_load_to_lds(g0, g1, z4, z4, z8, 0);
#else
  __builtin_amdgcn_tensor_load_to_lds(g0, g1, z4, z4, 0);
#endif
}

// ------------------------------ main kernel --------------------------------

__launch_bounds__(256)
__global__ void rpn_fused(const __bf16* __restrict__ Xp, const __bf16* __restrict__ Wt,
                          const __bf16* __restrict__ Wcr,
                          const float* __restrict__ hb, const float* __restrict__ cb,
                          const float* __restrict__ rb, float* __restrict__ out) {
  extern __shared__ __align__(16) char smem[];
  const int t    = threadIdx.x;
  const int lane = t & 31;
  // wave id is wave-uniform: pin it in an SGPR so wave-0 branches stay scalar
  const int wave = __builtin_amdgcn_readfirstlane(t >> 5);
  const int blk  = blockIdx.x;          // 256 blocks = (b, y)
  const int b = blk >> 5, y = blk & 31;

  const int wm    = wave >> 2;          // 0..1 : 16-pixel M-frag
  const int nbase = (wave & 3) << 7;    // 0,128,256,384

  const unsigned lds_base = (unsigned)(uintptr_t)smem;  // LDS aperture: low 32b

  v8f acc[8] = {};

  // issue both TDM tile loads for k-step s into buffer s&1 (wave 0 only)
  auto issue = [&](int s) {
    const int tap = s >> 6;
    const int k0  = (s & 63) << 5;
    const int ky  = tap / 3, kx = tap - ky * 3;
    const int buf = s & 1;
    tdm_load_tile(lds_base + AS0_OFF + buf * 2560,
                  Xp + ((size_t)((b * 34 + y + ky) * 34 + kx) << 11) + k0, 32);
    tdm_load_tile(lds_base + BS0_OFF + buf * 40960,
                  Wt + ((size_t)(tap * HID) << 11) + k0, 512);
  };

  if (wave == 0) issue(0);

  const int arow = (wm << 4) + (lane & 15);
  const int asel = (lane >> 4) << 4;
  const int bsel = (lane >> 4) << 4;
  // per-lane base byte offsets of the 8 B-fragment rows (within a B buffer)
  int boff[8];
#pragma unroll
  for (int j = 0; j < 8; ++j)
    boff[j] = (nbase + (j << 4) + (lane & 15)) * BS_STRIDE + bsel;

  for (int s = 0; s < STEPS; ++s) {
    if (wave == 0) {
      if (s + 1 < STEPS) {
        issue(s + 1);                                  // other buffer
        __builtin_amdgcn_s_wait_tensorcnt((short)2);   // step-s pair done
      } else {
        __builtin_amdgcn_s_wait_tensorcnt((short)0);
      }
    }
    __syncthreads();                                   // tile s visible to all

    const char* aP = smem + AS0_OFF + (s & 1) * 2560 + arow * AS_STRIDE + asel;
    const char* bB = smem + BS0_OFF + (s & 1) * 40960;
    v16bf fa = ld_frag(aP, aP + 32);

    // depth-2 pipelined B-fragment loads: load j+2 before WMMA j issues.
    v16bf f0 = ld_frag(bB + boff[0], bB + boff[0] + 32);
    v16bf f1 = ld_frag(bB + boff[1], bB + boff[1] + 32);
#pragma unroll
    for (int j = 0; j < 8; ++j) {
      v16bf cur = f0;
      f0 = f1;
      if (j + 2 < 8) f1 = ld_frag(bB + boff[j + 2], bB + boff[j + 2] + 32);
      acc[j] = __builtin_amdgcn_wmma_f32_16x16x32_bf16(
          false, fa, false, cur, (short)0, acc[j], false, false);
    }
    __syncthreads();   // all waves done with buffer s&1 before TDM reuses it
  }

  // ---- epilogue 1: bias + ReLU, hidden tile -> LDS (bf16), aliases B buf 0
#pragma unroll
  for (int j = 0; j < 8; ++j) {
    const int n = nbase + (j << 4) + (lane & 15);
    const float bias = hb[n];
#pragma unroll
    for (int r = 0; r < 8; ++r) {
      const int m = (wm << 4) + r + ((lane >> 4) << 3);
      float v = acc[j][r] + bias;
      v = fmaxf(v, 0.0f);
      *reinterpret_cast<__bf16*>(smem + H_OFF + m * H_STRIDE + n * 2) = (__bf16)v;
    }
  }
  __syncthreads();

  // ---- phase 2: [32 x 48] = H[32 x 512] * Wcr^T, 6 waves, WMMA
  if (wave < 6) {
    const int mf = (wave >= 3) ? 1 : 0;
    const int nf = wave - mf * 3;
    v8f a2 = {};
    const int hrow = (mf << 4) + (lane & 15);
    const int ncol = (nf << 4) + (lane & 15);
#pragma unroll
    for (int kk = 0; kk < 16; ++kk) {
      const int k0 = kk << 5;
      const char* hp = smem + H_OFF + hrow * H_STRIDE + (k0 << 1) + asel;
      v16bf fa = ld_frag(hp, hp + 32);
      const __bf16* wp = Wcr + ncol * HID + k0 + (asel >> 1);
      v16bf fw = ld_frag(wp, wp + 16);
      a2 = __builtin_amdgcn_wmma_f32_16x16x32_bf16(
          false, fa, false, fw, (short)0, a2, false, false);
    }
#pragma unroll
    for (int r = 0; r < 8; ++r) {
      const int m = (mf << 4) + r + ((lane >> 4) << 3);
      *reinterpret_cast<float*>(smem + S_OFF + (m * S_STRIDE + ncol) * 4) = a2[r];
    }
  }
  __syncthreads();

  // ---- phase 3: sigmoid / exp / anchor decode, write [B, A, 5]
  if (t < 32 * 9) {
    const int p = t / 9;              // pixel x
    const int a = t - p * 9;          // anchor
    const float* Sp = reinterpret_cast<const float*>(smem + S_OFF) + p * S_STRIDE;
    const float conf = Sp[a] + cb[a];
    const float t0 = Sp[9 + a * 4 + 0] + rb[a * 4 + 0];
    const float t1 = Sp[9 + a * 4 + 1] + rb[a * 4 + 1];
    const float t2 = Sp[9 + a * 4 + 2] + rb[a * 4 + 2];
    const float t3 = Sp[9 + a * 4 + 3] + rb[a * 4 + 3];

    const float sc = (a < 3) ? 2.0f : ((a < 6) ? 4.0f : 6.0f);
    const int rm = a - (a / 3) * 3;
    const float rt = (rm == 0) ? 0.5f : ((rm == 1) ? 1.0f : 1.5f);
    const float wA = sc * rt, hA = sc;
    const float cx = p + 0.5f, cy = y + 0.5f;
    const float x1 = fminf(fmaxf(cx - wA * 0.5f, 0.0f), 32.0f);
    const float y1 = fminf(fmaxf(cy - hA * 0.5f, 0.0f), 32.0f);
    const float x2 = fminf(fmaxf(cx + wA * 0.5f, 0.0f), 32.0f);
    const float y2 = fminf(fmaxf(cy + hA * 0.5f, 0.0f), 32.0f);
    const float aw = x2 - x1, ah = y2 - y1;
    const float acx = x1 + aw * 0.5f, acy = y1 + ah * 0.5f;
    const float pcx = acx + t0 * aw, pcy = acy + t1 * ah;
    const float pw = aw * __expf(t2), ph = ah * __expf(t3);
    const float score = 1.0f / (1.0f + __expf(-conf));

    const size_t o = ((((size_t)b * 1024 + y * 32 + p) * 9) + a) * 5;
    out[o + 0] = score;
    out[o + 1] = (pcx - pw * 0.5f) * 32.0f;
    out[o + 2] = (pcy - ph * 0.5f) * 32.0f;
    out[o + 3] = (pcx + pw * 0.5f) * 32.0f;
    out[o + 4] = (pcy + ph * 0.5f) * 32.0f;
  }
}

// ------------------------------- launcher ----------------------------------

extern "C" void kernel_launch(void* const* d_in, const int* in_sizes, int n_in,
                              void* d_out, int out_size, void* d_ws, size_t ws_size,
                              hipStream_t stream) {
  (void)in_sizes; (void)n_in; (void)out_size; (void)ws_size;
  const float* fm = (const float*)d_in[0];
  const float* hw = (const float*)d_in[1];
  const float* hb = (const float*)d_in[2];
  const float* cw = (const float*)d_in[3];
  const float* cb = (const float*)d_in[4];
  const float* rw = (const float*)d_in[5];
  const float* rb = (const float*)d_in[6];

  char* ws = (char*)d_ws;
  __bf16* Xp  = (__bf16*)(ws);
  __bf16* Wt  = (__bf16*)(ws + WT_OFF);
  __bf16* Wcr = (__bf16*)(ws + WCR_OFF);

  {
    const int nu32 = (int)(XP_BYTES / 4);
    k_zero_u32<<<nu32 / 256, 256, 0, stream>>>((unsigned int*)Xp, nu32);
  }
  k_pad_nhwc<<<256, 256, 0, stream>>>(fm, Xp);
  k_pack_wt<<<(9 * 512 * 2048) / 256, 256, 0, stream>>>(hw, Wt);
  k_pack_wcr<<<(48 * 512) / 256, 256, 0, stream>>>(cw, rw, Wcr);

  rpn_fused<<<256, 256, DYN_LDS, stream>>>(Xp, Wt, Wcr, hb, cb, rb, (float*)d_out);
}
